// SelfAttention_82815559401983
// MI455X (gfx1250) — matrix operand order repaired and checked by reading, not compiled, not run on hardware
//
#include <hip/hip_runtime.h>
#include <hip/hip_bf16.h>
#include <math.h>

// ---------------------------------------------------------------------------
// Self-attention (B=2, S=4096, D=512, H=8, dh=64) for gfx1250 (MI455X).
// f16 WMMA (16x16x32, f32 accumulate); flash attention with double-buffered
// async global->LDS staging of K/V tiles (ASYNCcnt) shared by 4 waves.
// ---------------------------------------------------------------------------

typedef __attribute__((ext_vector_type(16))) _Float16 v16h;
typedef __attribute__((ext_vector_type(8)))  _Float16 v8h;
typedef __attribute__((ext_vector_type(8)))  float    v8f;
typedef __attribute__((ext_vector_type(4)))  float    v4f;

#define D_MODEL 512
#define NHEAD   8
#define DHEAD   64
#define BATCH   2
#define SEQ     4096
#define MROWS   (BATCH * SEQ)   // 8192

static __device__ __forceinline__ v16h cat16(v8h lo, v8h hi) {
  v16h r;
#pragma unroll
  for (int i = 0; i < 8; ++i) { r[i] = lo[i]; r[i + 8] = hi[i]; }
  return r;
}

static __device__ __forceinline__ v8f wmma16(v16h a, v16h b, v8f c) {
  return __builtin_amdgcn_wmma_f32_16x16x32_f16(false, a, false, b,
                                                (short)0, c, false, false);
}

// Async global->LDS copy, GVS addressing (lds-addr VGPR, u32 voffset VGPR,
// u64 SGPR base). Tracked by ASYNCcnt.
#define ASYNC_LOAD_B128(ldsaddr, voff, base)                        \
  asm volatile("global_load_async_to_lds_b128 %0, %1, %2"           \
               :: "v"(ldsaddr), "v"(voff), "s"(base) : "memory")

#if __has_builtin(__builtin_amdgcn_s_wait_asynccnt)
#define WAIT_ASYNC(n) __builtin_amdgcn_s_wait_asynccnt(n)
#else
#define WAIT_ASYNC(n) asm volatile("s_wait_asynccnt " #n ::: "memory")
#endif

// ---------------------------------------------------------------------------
// Weight transpose + f32->f16 convert:  WT[n][k] = (f16) W[k][n]
// ---------------------------------------------------------------------------
__global__ __launch_bounds__(256) void k_transpose_f16(
    const float* __restrict__ W, _Float16* __restrict__ WT) {
  int idx = blockIdx.x * 256 + threadIdx.x;   // over 512*512
  int n = idx >> 9;
  int k = idx & 511;
  WT[idx] = (_Float16)W[((size_t)k << 9) + n];
}

// ---------------------------------------------------------------------------
// Projection GEMM: Y = x @ W + b -> f16 head-split layout.
//   mode 0: Q -> [bh][s][d], pre-scaled by 1/sqrt(dh)=0.125
//   mode 1: K -> [bh][s][d]
//   mode 2: V -> [bh][d][s] (transposed)
// One wave = 16x64 output tile. All 4 B fragments loaded up-front per k-step
// so WMMAs retire against partial load counts instead of waiting one-by-one.
// ---------------------------------------------------------------------------
__global__ __launch_bounds__(256) void k_proj(
    const float* __restrict__ x, const _Float16* __restrict__ WT,
    const float* __restrict__ bias, _Float16* __restrict__ out, int mode) {
  const int lane = threadIdx.x & 31;
  const int wave = threadIdx.x >> 5;
  const int tile = blockIdx.x * 8 + wave;       // 4096 waves total
  const int m0   = (tile >> 3) << 4;            // 512 row tiles
  const int n0   = (tile & 7) << 6;             // 8 col groups of 64
  const int lm   = lane & 15;
  const int hid  = lane >> 4;

  const float*    xrow  = x  + (size_t)(m0 + lm) * D_MODEL;
  const _Float16* wbase = WT + (size_t)(n0 + lm) * D_MODEL;

  v8f acc[4];
#pragma unroll
  for (int t = 0; t < 4; ++t) acc[t] = (v8f){};

#pragma unroll 2
  for (int k0 = 0; k0 < D_MODEL; k0 += 32) {
    // Issue all loads first: 4 fp32x4 for A, 8 b128 for the 4 B fragments.
    const int abo = k0 + hid * 8;
    v4f a0 = *(const v4f*)(xrow + abo);
    v4f a1 = *(const v4f*)(xrow + abo + 4);
    v4f a2 = *(const v4f*)(xrow + abo + 16);
    v4f a3 = *(const v4f*)(xrow + abo + 20);

    const int bb = k0 + hid * 8;
    v16h bfr[4];
#pragma unroll
    for (int t = 0; t < 4; ++t) {
      const _Float16* wrow = wbase + (size_t)(t * 16) * D_MODEL + bb;
      bfr[t] = cat16(*(const v8h*)(wrow), *(const v8h*)(wrow + 16));
    }

    v16h a;
#pragma unroll
    for (int i = 0; i < 4; ++i) {
      a[i]      = (_Float16)a0[i];
      a[i + 4]  = (_Float16)a1[i];
      a[i + 8]  = (_Float16)a2[i];
      a[i + 12] = (_Float16)a3[i];
    }
#pragma unroll
    for (int t = 0; t < 4; ++t) acc[t] = wmma16(a, bfr[t], acc[t]);
  }

  const int h = n0 >> 6;   // head (n0 is a multiple of 64)
  if (mode == 2) {
    const int gm0 = m0 + hid * 8;
    const int bb_ = gm0 >> 12;
    const int sb  = gm0 & (SEQ - 1);
#pragma unroll
    for (int t = 0; t < 4; ++t) {
      const int d = t * 16 + lm;
      const float bv = bias[n0 + d];
      v8h pk;
#pragma unroll
      for (int r = 0; r < 8; ++r) pk[r] = (_Float16)(acc[t][r] + bv);
      *(v8h*)(out + ((size_t)(bb_ * NHEAD + h) * DHEAD + d) * SEQ + sb) = pk;
    }
  } else {
    const float scale = (mode == 0) ? 0.125f : 1.0f;
#pragma unroll
    for (int t = 0; t < 4; ++t) {
      const int d = t * 16 + lm;
      const float bv = bias[n0 + d];
#pragma unroll
      for (int r = 0; r < 8; ++r) {
        const int gm = m0 + r + hid * 8;
        const int bb_ = gm >> 12;
        const int s   = gm & (SEQ - 1);
        out[((size_t)(bb_ * NHEAD + h) * SEQ + s) * DHEAD + d] =
            (_Float16)((acc[t][r] + bv) * scale);
      }
    }
  }
}

// ---------------------------------------------------------------------------
// Flash attention. Block = 128 threads (4 waves) = 64 query rows.
// Per 64-key tile: K (8KB) + V (8KB) staged in LDS via async copies
// (double-buffered), shared by all 4 waves. 16 WMMA per wave per tile.
// ---------------------------------------------------------------------------
// smem (halfs): K[2][64][64] | V[2][64][64] | P[4][16][64]
#define SM_K(b)   ((b) * 4096)
#define SM_V(b)   (8192 + (b) * 4096)
#define SM_P(w)   (16384 + (w) * 1024)
#define SM_TOTAL  (16384 + 4096)

static __device__ __forceinline__ void issue_tile(
    const _Float16* Kp, const _Float16* Vp, _Float16* sm,
    int jn, int bufn, int tid) {
  const uint64_t kb = (uint64_t)(uintptr_t)Kp;
  const uint64_t vb = (uint64_t)(uintptr_t)Vp;
#pragma unroll
  for (int i = 0; i < 4; ++i) {        // K tile: contiguous 8KB
    const int c = tid + i * 128;       // 16B chunk id, [0,512)
    const unsigned la = (unsigned)(uintptr_t)(sm + SM_K(bufn) + c * 8);
    const unsigned vo = (unsigned)(jn * 128 + c * 16);
    ASYNC_LOAD_B128(la, vo, kb);
  }
#pragma unroll
  for (int i = 0; i < 4; ++i) {        // V tile: 64 rows x 128B (stride SEQ)
    const int c = tid + i * 128;
    const int d = c >> 3, part = c & 7;
    const unsigned la = (unsigned)(uintptr_t)(sm + SM_V(bufn) + d * 64 + part * 8);
    const unsigned vo = ((unsigned)d * SEQ + (unsigned)jn + part * 8) * 2;
    ASYNC_LOAD_B128(la, vo, vb);
  }
}

__global__ __launch_bounds__(128) void k_flash_attn(
    const _Float16* __restrict__ Qh, const _Float16* __restrict__ Kh,
    const _Float16* __restrict__ Vt, _Float16* __restrict__ Oh) {
  __shared__ _Float16 smem[SM_TOTAL];

  const int tid  = threadIdx.x;
  const int lane = tid & 31;
  const int w    = tid >> 5;
  const int bh   = blockIdx.x >> 6;
  const int qblk = blockIdx.x & 63;
  const int lm   = lane & 15;
  const int hid  = lane >> 4;
  const int qs0  = qblk * 64 + w * 16;

  const _Float16* Qp = Qh + (size_t)bh * SEQ * DHEAD;
  const _Float16* Kp = Kh + (size_t)bh * SEQ * DHEAD;
  const _Float16* Vp = Vt + (size_t)bh * DHEAD * SEQ;

  // Q A-fragments held in registers all pass (Q pre-scaled by 0.125).
  const _Float16* qrow = Qp + (size_t)(qs0 + lm) * DHEAD;
  const int ab = hid * 8;
  const v16h qa0 = cat16(*(const v8h*)(qrow + ab),      *(const v8h*)(qrow + ab + 16));
  const v16h qa1 = cat16(*(const v8h*)(qrow + ab + 32), *(const v8h*)(qrow + ab + 48));

  float m[8], l[8];
  v8f acc[4];
#pragma unroll
  for (int r = 0; r < 8; ++r) { m[r] = -__builtin_inff(); l[r] = 0.0f; }
#pragma unroll
  for (int t = 0; t < 4; ++t) acc[t] = (v8f){};

  issue_tile(Kp, Vp, smem, 0, 0, tid);          // prefetch first tile

  for (int j = 0; j < SEQ; j += 64) {
    const int buf = (j >> 6) & 1;
    const bool more = (j + 64) < SEQ;
    if (more) {
      issue_tile(Kp, Vp, smem, j + 64, buf ^ 1, tid);  // prefetch next
      WAIT_ASYNC(8);                                   // current tile done
    } else {
      WAIT_ASYNC(0);
    }
    __syncthreads();                                   // tile visible to all

    const _Float16* Kl = smem + SM_K(buf);
    const _Float16* Vl = smem + SM_V(buf);
    _Float16*       Pl = smem + SM_P(w);

    // ---- scores: S = Q(16x64) . K^T(64x16) per 16-key subtile ----
    v8f sc[4];
#pragma unroll
    for (int t = 0; t < 4; ++t) {
      const _Float16* krow = Kl + (t * 16 + lm) * 64;
      const int kb = hid * 8;
      v16h b0 = cat16(*(const v8h*)(krow + kb),      *(const v8h*)(krow + kb + 16));
      v16h b1 = cat16(*(const v8h*)(krow + 32 + kb), *(const v8h*)(krow + 32 + kb + 16));
      v8f s = {};
      s = wmma16(qa0, b0, s);
      s = wmma16(qa1, b1, s);
      sc[t] = s;
    }

    // ---- online softmax (row stats across 16-lane half-groups) ----
#pragma unroll
    for (int r = 0; r < 8; ++r) {
      float cm = fmaxf(fmaxf(sc[0][r], sc[1][r]), fmaxf(sc[2][r], sc[3][r]));
      cm = fmaxf(cm, __shfl_xor(cm, 1, 32));
      cm = fmaxf(cm, __shfl_xor(cm, 2, 32));
      cm = fmaxf(cm, __shfl_xor(cm, 4, 32));
      cm = fmaxf(cm, __shfl_xor(cm, 8, 32));
      const float mnew  = fmaxf(m[r], cm);
      const float alpha = __expf(m[r] - mnew);
      float p0 = __expf(sc[0][r] - mnew);
      float p1 = __expf(sc[1][r] - mnew);
      float p2 = __expf(sc[2][r] - mnew);
      float p3 = __expf(sc[3][r] - mnew);
      float ps = (p0 + p1) + (p2 + p3);
      ps += __shfl_xor(ps, 1, 32);
      ps += __shfl_xor(ps, 2, 32);
      ps += __shfl_xor(ps, 4, 32);
      ps += __shfl_xor(ps, 8, 32);
      l[r] = l[r] * alpha + ps;
      m[r] = mnew;
#pragma unroll
      for (int t = 0; t < 4; ++t) acc[t][r] *= alpha;
      const int row = r + hid * 8;
      Pl[row * 64 +  0 + lm] = (_Float16)p0;
      Pl[row * 64 + 16 + lm] = (_Float16)p1;
      Pl[row * 64 + 32 + lm] = (_Float16)p2;
      Pl[row * 64 + 48 + lm] = (_Float16)p3;
    }
    __syncthreads();                    // P relayout (C-layout -> A-layout)

    const _Float16* prow = Pl + lm * 64;
    v16h pa0 = cat16(*(const v8h*)(prow + ab),      *(const v8h*)(prow + ab + 16));
    v16h pa1 = cat16(*(const v8h*)(prow + ab + 32), *(const v8h*)(prow + ab + 48));

    // ---- O += P(16x64) . V(64x64); V staged transposed [d][key] ----
#pragma unroll
    for (int t = 0; t < 4; ++t) {
      const _Float16* vrow = Vl + (t * 16 + lm) * 64;
      const int kb = hid * 8;
      v16h vb0 = cat16(*(const v8h*)(vrow + kb),      *(const v8h*)(vrow + kb + 16));
      v16h vb1 = cat16(*(const v8h*)(vrow + 32 + kb), *(const v8h*)(vrow + 32 + kb + 16));
      acc[t] = wmma16(pa0, vb0, acc[t]);
      acc[t] = wmma16(pa1, vb1, acc[t]);
    }
    __syncthreads();   // all reads of this buffer + P done -> safe to refill
  }

  // ---- epilogue: normalize, write merged-head f16 [b][s][h*64+d] ----
  const int b = bh >> 3;
  const int h = bh & 7;
#pragma unroll
  for (int r = 0; r < 8; ++r) {
    const float inv = 1.0f / l[r];
    const int s = qs0 + r + hid * 8;
    _Float16* orow = Oh + ((size_t)(b * SEQ + s)) * D_MODEL + h * DHEAD;
#pragma unroll
    for (int t = 0; t < 4; ++t)
      orow[t * 16 + lm] = (_Float16)(acc[t][r] * inv);
  }
}

// ---------------------------------------------------------------------------
// Output projection: out = Oh(f16) @ Wo + bo, fp32 result (16x64 per wave).
// ---------------------------------------------------------------------------
__global__ __launch_bounds__(256) void k_outproj(
    const _Float16* __restrict__ Oh, const _Float16* __restrict__ WoT,
    const float* __restrict__ bias, float* __restrict__ out) {
  const int lane = threadIdx.x & 31;
  const int wave = threadIdx.x >> 5;
  const int tile = blockIdx.x * 8 + wave;
  const int m0   = (tile >> 3) << 4;
  const int n0   = (tile & 7) << 6;
  const int lm   = lane & 15;
  const int hid  = lane >> 4;

  const _Float16* arow  = Oh  + (size_t)(m0 + lm) * D_MODEL;
  const _Float16* wbase = WoT + (size_t)(n0 + lm) * D_MODEL;

  v8f acc[4];
#pragma unroll
  for (int t = 0; t < 4; ++t) acc[t] = (v8f){};

#pragma unroll 2
  for (int k0 = 0; k0 < D_MODEL; k0 += 32) {
    const int abo = k0 + hid * 8;
    v8h alo = *(const v8h*)(arow + abo);
    v8h ahi = *(const v8h*)(arow + abo + 16);

    const int bb = k0 + hid * 8;
    v16h bfr[4];
#pragma unroll
    for (int t = 0; t < 4; ++t) {
      const _Float16* wrow = wbase + (size_t)(t * 16) * D_MODEL + bb;
      bfr[t] = cat16(*(const v8h*)(wrow), *(const v8h*)(wrow + 16));
    }

    v16h a = cat16(alo, ahi);
#pragma unroll
    for (int t = 0; t < 4; ++t) acc[t] = wmma16(a, bfr[t], acc[t]);
  }

#pragma unroll
  for (int t = 0; t < 4; ++t) {
    const int n = n0 + t * 16 + lm;
    const float bv = bias[n];
#pragma unroll
    for (int r = 0; r < 8; ++r) {
      const int gm = m0 + r + hid * 8;
      out[(size_t)gm * D_MODEL + n] = acc[t][r] + bv;
    }
  }
}

// ---------------------------------------------------------------------------
extern "C" void kernel_launch(void* const* d_in, const int* in_sizes, int n_in,
                              void* d_out, int out_size, void* d_ws, size_t ws_size,
                              hipStream_t stream) {
  (void)in_sizes; (void)n_in; (void)out_size; (void)ws_size;
  const float* x  = (const float*)d_in[0];
  const float* Wq = (const float*)d_in[1];
  const float* bq = (const float*)d_in[2];
  const float* Wk = (const float*)d_in[3];
  const float* bk = (const float*)d_in[4];
  const float* Wv = (const float*)d_in[5];
  const float* bv = (const float*)d_in[6];
  const float* Wo = (const float*)d_in[7];
  const float* bo = (const float*)d_in[8];
  float* out = (float*)d_out;

  _Float16* WTq = (_Float16*)d_ws;
  _Float16* WTk = WTq + (size_t)D_MODEL * D_MODEL;
  _Float16* WTv = WTk + (size_t)D_MODEL * D_MODEL;
  _Float16* WTo = WTv + (size_t)D_MODEL * D_MODEL;
  _Float16* Qh  = WTo + (size_t)D_MODEL * D_MODEL;
  _Float16* Kh  = Qh  + (size_t)MROWS * D_MODEL;
  _Float16* Vt  = Kh  + (size_t)MROWS * D_MODEL;
  _Float16* Oh  = Vt  + (size_t)MROWS * D_MODEL;

  k_transpose_f16<<<1024, 256, 0, stream>>>(Wq, WTq);
  k_transpose_f16<<<1024, 256, 0, stream>>>(Wk, WTk);
  k_transpose_f16<<<1024, 256, 0, stream>>>(Wv, WTv);
  k_transpose_f16<<<1024, 256, 0, stream>>>(Wo, WTo);

  k_proj<<<512, 256, 0, stream>>>(x, WTq, bq, Qh, 0);   // Q (pre-scaled)
  k_proj<<<512, 256, 0, stream>>>(x, WTk, bk, Kh, 1);   // K
  k_proj<<<512, 256, 0, stream>>>(x, WTv, bv, Vt, 2);   // V (transposed)

  k_flash_attn<<<BATCH * NHEAD * (SEQ / 64), 128, 0, stream>>>(Qh, Kh, Vt, Oh);

  k_outproj<<<512, 256, 0, stream>>>(Oh, WTo, bo, out);
}
